// FastHyperbolicAttention_7576322310246
// MI455X (gfx1250) — compile-verified, matrix operand order checked by reading, not run on hardware
//
#include <hip/hip_runtime.h>
#include <math.h>

// ---------------------------------------------------------------------------
// Hyperbolic (Poincare) attention for MI455X / gfx1250, wave32 + WMMA + TDM.
//
// Roofline: weights output (B,H,L,L)=1.07GB fp32 -> ~46us HBM floor.
// All activations (~150MB) fit in 192MB L2. Projection GEMMs run on
// v_wmma_f32_16x16x32_f16 (f16 in / f32 acc); the softmax is ~argmax of
// the Poincare distance (conf terms ~2e-5 => scores ~ -1e10*diff^2), so
// the distance + weights@V path stays fp32 on v_wmma_f32_16x16x4_f32.
//
// Attention K/V tiles are staged into LDS by the Tensor Data Mover
// (tensor_load_to_lds, double-buffered, s_wait_tensorcnt) and shared by
// all 8 waves of a block (which provably share the same (b,h)).
// TDM padding (4 dwords per 64-dword row -> pitch 68 floats) makes the
// half-wave striped ds_load_b64 operand fetch bank-conflict-free.
// sched_barrier splits operand-gather from the WMMA burst so the LDS
// round-trip is exposed once per tile, not once per WMMA pair.
// ---------------------------------------------------------------------------

#define EPSF     1e-6f
#define MAXNORM  (1.0f - 1e-5f)
#define NEG_BIG  (-3.0e38f)

constexpr int Bb   = 4;
constexpr int Ls   = 2048;
constexpr int Dd   = 1024;
constexpr int Hh   = 16;
constexpr int HD   = 64;
constexpr int MTOT = Bb * Ls;        // 8192
constexpr int KT   = Ls / 16;        // 128 k-tiles per row
constexpr int KP   = 68;             // padded LDS row pitch for K/V (floats)
constexpr int WP   = 20;             // w-tile pitch (floats, 16B-aligned rows)

typedef __attribute__((ext_vector_type(16))) _Float16 v16h;
typedef __attribute__((ext_vector_type(8)))  _Float16 v8h;
typedef __attribute__((ext_vector_type(8)))  float    v8f;
typedef __attribute__((ext_vector_type(2)))  float    v2f;
typedef __attribute__((ext_vector_type(4)))  unsigned int v4u;
typedef __attribute__((ext_vector_type(8)))  int      v8i;
typedef __attribute__((ext_vector_type(4)))  int      v4i;

#if defined(__has_builtin)
#if __has_builtin(__builtin_amdgcn_tensor_load_to_lds)
#define HAVE_TDM 1
#else
#define HAVE_TDM 0
#endif
#if __has_builtin(__builtin_amdgcn_sched_barrier)
#define SCHED_FENCE() __builtin_amdgcn_sched_barrier(0)
#else
#define SCHED_FENCE()
#endif
#else
#define HAVE_TDM 0
#define SCHED_FENCE()
#endif

static __device__ inline v8f zero8() {
  v8f z;
#pragma unroll
  for (int i = 0; i < 8; ++i) z[i] = 0.0f;
  return z;
}

// Build a 16xf16 A/B operand from two contiguous 16B chunks
// (lane's K-pairs: [k0+8*half .. +7] and [k0+16+8*half .. +7]).
static __device__ inline v16h load_ab16(const _Float16* p) {
  v8h lo = *(const v8h*)(p);
  v8h hi = *(const v8h*)(p + 16);
  v16h r;
#pragma unroll
  for (int i = 0; i < 8; ++i) { r[i] = lo[i]; r[i + 8] = hi[i]; }
  return r;
}

// ------------------- TDM staging of a 16x64 f32 tile -----------------------
// D# per CDNA5 ISA: group0 = {count|flags, lds_addr,
// global_addr}, group1 = {mask/data_size/pad, tensor dims, tile dims, stride}.
// pad: 4 dwords every 64 dwords -> LDS row pitch = 68 floats (bank-spread).
static __device__ inline void stage_tile_16x64_f32(float* __restrict__ dst,
                                                   const float* __restrict__ src) {
#if HAVE_TDM
  if (threadIdx.x < 32) {   // one TDM op per block (EXEC ignored by TDM)
    unsigned long long ga = (unsigned long long)(uintptr_t)src;
    v4u g0;
    g0[0] = 1u;                                        // count=1, user D#
    g0[1] = (unsigned)(uintptr_t)dst;                  // LDS byte offset
    g0[2] = (unsigned)ga;                              // global_addr[31:0]
    g0[3] = (unsigned)((ga >> 32) & 0x1ffffffu) | (2u << 30);  // addr hi | type=2
    v8i g1;
    g1[0] = (int)((2u << 16) |      // data_size = 4B
                  (1u << 20) |      // pad_enable
                  (5u << 22) |      // pad_interval: every 64 dwords
                  (3u << 25));      // pad_amount: 4 dwords
    g1[1] = (int)(64u << 16);       // tensor_dim0 = 64       (bits 79:48 lo)
    g1[2] = (int)(2048u << 16);     // tensor_dim1 = 2048     (bits 111:80 lo)
    g1[3] = (int)(64u << 16);       // tile_dim0 = 64         (bits 127:112)
    g1[4] = 16;                     // tile_dim1 = 16, tile_dim2 = 0
    g1[5] = 64;                     // tensor_dim0_stride = 64 elems
    g1[6] = 0;
    g1[7] = 0;
    v4i z4 = {0, 0, 0, 0};
#if __clang_major__ >= 23
    v8i z8 = {0, 0, 0, 0, 0, 0, 0, 0};
    __builtin_amdgcn_tensor_load_to_lds(g0, g1, z4, z4, z8, 0);
#else
    __builtin_amdgcn_tensor_load_to_lds(g0, g1, z4, z4, 0);
#endif
  }
#else
  // Cooperative fallback: 256 threads copy 16 rows x 64 floats with pitch 68.
  const int t = threadIdx.x;
  const int r = t >> 4, c = (t & 15) * 4;
  *(float4*)(dst + r * KP + c) = *(const float4*)(src + r * 64 + c);
#endif
}

template <int N>
static __device__ inline void stage_wait() {
#if HAVE_TDM
  if (threadIdx.x < 32) __builtin_amdgcn_s_wait_tensorcnt((short)N);
#endif
}

// --------------------------- f32 -> f16 convert ----------------------------
__global__ void hypatt_cvt_f16(const float* __restrict__ x,
                               _Float16* __restrict__ y, int n) {
  int i = blockIdx.x * blockDim.x + threadIdx.x;
  if (i < n) y[i] = (_Float16)x[i];
}

// ------------------------------ GEMM (f16) ---------------------------------
// Y = X @ W^T + b.  One wave computes a 16(M) x 64(N) tile; block = 8 waves.
// MODE 0: plain (MTOT,Dd) output.
// MODE 1: scatter to (B,H,L,64).
// MODE 2: scatter + Poincare-ball projection + post-projection |x|^2 stats.
template <int MODE>
__global__ __launch_bounds__(256) void hypatt_gemm(
    const _Float16* __restrict__ A,   // (MTOT, Dd) f16 row-major
    const _Float16* __restrict__ W,   // (Dd, Dd)  f16 row-major (use W^T)
    const float* __restrict__ bias,   // (Dd)
    float* __restrict__ out,
    float* __restrict__ sq)           // (B*H*L) or nullptr
{
  const int wib  = threadIdx.x >> 5;
  const int wid  = blockIdx.x * 8 + wib;            // 0..8191
  const int lane = threadIdx.x & 31;
  const int hf   = lane >> 4;
  const int lr   = lane & 15;
  const int mt   = wid >> 4;                        // 0..511
  const int nt   = wid & 15;                        // head / 64-col group

  const _Float16* arow = A + (size_t)(mt * 16 + lr) * Dd + 8 * hf;
  const _Float16* brow = W + (size_t)(nt * 64 + lr) * Dd + 8 * hf;

  v8f acc[4];
#pragma unroll
  for (int s = 0; s < 4; ++s) acc[s] = zero8();

  for (int k0 = 0; k0 < Dd; k0 += 32) {
    __builtin_prefetch(arow + k0 + 128, 0, 0);
    v16h a = load_ab16(arow + k0);
    v16h b[4];
#pragma unroll
    for (int sub = 0; sub < 4; ++sub)
      b[sub] = load_ab16(brow + (size_t)sub * 16 * Dd + k0);
#pragma unroll
    for (int sub = 0; sub < 4; ++sub)
      acc[sub] = __builtin_amdgcn_wmma_f32_16x16x32_f16(
          false, a, false, b[sub], (short)0, acc[sub], false, false);
  }

  // bias (C layout: N = lane&15, same for both halves)
#pragma unroll
  for (int sub = 0; sub < 4; ++sub) {
    float bn = bias[nt * 64 + sub * 16 + lr];
#pragma unroll
    for (int j = 0; j < 8; ++j) acc[sub][j] += bn;
  }

  if (MODE == 2) {
    // hd == 64 == wave tile width: full per-row norm lives in this wave.
#pragma unroll
    for (int j = 0; j < 8; ++j) {
      float p = acc[0][j] * acc[0][j] + acc[1][j] * acc[1][j] +
                acc[2][j] * acc[2][j] + acc[3][j] * acc[3][j];
#pragma unroll
      for (int off = 8; off; off >>= 1) p += __shfl_xor(p, off, 16);
      float nrm = fmaxf(sqrtf(p), EPSF);
      float s   = (nrm < MAXNORM) ? 1.0f : (MAXNORM / nrm);
#pragma unroll
      for (int sub = 0; sub < 4; ++sub) acc[sub][j] *= s;
      if (lr == 0) {
        int mg = mt * 16 + j + 8 * hf;
        int b  = mg >> 11;
        int l  = mg & (Ls - 1);
        sq[(size_t)(b * Hh + nt) * Ls + l] = p * s * s;
      }
    }
  }

#pragma unroll
  for (int j = 0; j < 8; ++j) {
    int mg = mt * 16 + j + 8 * hf;
    if (MODE == 0) {
#pragma unroll
      for (int sub = 0; sub < 4; ++sub)
        out[(size_t)mg * Dd + nt * 64 + sub * 16 + lr] = acc[sub][j];
    } else {
      int b = mg >> 11;
      int l = mg & (Ls - 1);
      float* o = out + ((size_t)(b * Hh + nt) * Ls + l) * HD + lr;
#pragma unroll
      for (int sub = 0; sub < 4; ++sub) o[sub * 16] = acc[sub][j];
    }
  }
}

// --------------------- attention pass 1: rowmax/rowsum ---------------------
// One wave per 16-row q-tile; 8 waves/block share (b,h). K tiles arrive via
// TDM double-buffer in LDS. Per-lane online softmax, half-wave combine.
// s = dsq * (-4/cx) * (1/cy): -4/cx hoisted per row, 1/cy once per tile.
__global__ __launch_bounds__(256) void hypatt_stats(
    const float* __restrict__ Qp, const float* __restrict__ Kp,
    const float* __restrict__ QSQ, const float* __restrict__ KSQ,
    float* __restrict__ rowM, float* __restrict__ rowS)
{
  __shared__ float kbuf[2][16 * KP];

  const int wib  = threadIdx.x >> 5;
  const int wid  = blockIdx.x * 8 + wib;  // 0..8191
  const int lane = threadIdx.x & 31;
  const int hf   = lane >> 4;
  const int lr   = lane & 15;
  const int bh   = wid >> 7;              // uniform across block
  const int qt   = wid & 127;

  const float* Qb  = Qp  + (size_t)bh * Ls * HD;
  const float* Kb  = Kp  + (size_t)bh * Ls * HD;
  const float* QSb = QSQ + (size_t)bh * Ls;
  const float* KSb = KSQ + (size_t)bh * Ls;

  v2f qa[16];
  const float* qrow = Qb + (size_t)(qt * 16 + lr) * HD + 2 * hf;
#pragma unroll
  for (int i = 0; i < 16; ++i) qa[i] = *(const v2f*)(qrow + 4 * i);

  float qs[8], icx[8];
#pragma unroll
  for (int j = 0; j < 8; ++j) {
    qs[j]  = QSb[qt * 16 + j + 8 * hf];
    icx[j] = -4.0f / fmaxf(1.0f - qs[j], EPSF);
  }

  float ml[8], sl[8];
#pragma unroll
  for (int j = 0; j < 8; ++j) { ml[j] = NEG_BIG; sl[j] = 0.0f; }

  stage_tile_16x64_f32(&kbuf[0][0], Kb);

  for (int kt = 0; kt < KT; ++kt) {
    if (kt + 1 < KT) {
      stage_tile_16x64_f32(&kbuf[(kt + 1) & 1][0], Kb + (size_t)(kt + 1) * 16 * HD);
      stage_wait<1>();
    } else {
      stage_wait<0>();
    }
    __syncthreads();

    // -------- operand gather phase (kept apart from the WMMA burst) --------
    float ks  = KSb[kt * 16 + lr];
    const float* krow = &kbuf[kt & 1][lr * KP + 2 * hf];
    v2f bb[16];
#pragma unroll
    for (int i = 0; i < 16; ++i) bb[i] = *(const v2f*)(krow + 4 * i);
    SCHED_FENCE();

    // -------- two independent WMMA chains halve the RAW critical path ------
    v8f xy0 = zero8(), xy1 = zero8();
#pragma unroll
    for (int i = 0; i < 8; ++i) {
      xy0 = __builtin_amdgcn_wmma_f32_16x16x4_f32(
          false, qa[2 * i],     false, bb[2 * i],     (short)0, xy0, false, false);
      xy1 = __builtin_amdgcn_wmma_f32_16x16x4_f32(
          false, qa[2 * i + 1], false, bb[2 * i + 1], (short)0, xy1, false, false);
    }

    float icy = 1.0f / fmaxf(1.0f - ks, EPSF);
#pragma unroll
    for (int j = 0; j < 8; ++j) {
      float xy  = xy0[j] + xy1[j];
      float dsq = fmaxf(qs[j] + ks - 2.0f * xy, 0.0f);
      float s   = dsq * icx[j] * icy;
      float nm  = fmaxf(ml[j], s);
      sl[j] = sl[j] * __expf(ml[j] - nm) + __expf(s - nm);
      ml[j] = nm;
    }
    __syncthreads();
  }

#pragma unroll
  for (int j = 0; j < 8; ++j) {
    float M = ml[j];
#pragma unroll
    for (int off = 8; off; off >>= 1) M = fmaxf(M, __shfl_xor(M, off, 16));
    float S = sl[j] * __expf(ml[j] - M);
#pragma unroll
    for (int off = 8; off; off >>= 1) S += __shfl_xor(S, off, 16);
    if (lr == 0) {
      int q = qt * 16 + j + 8 * hf;
      rowM[(size_t)bh * Ls + q] = M;
      rowS[(size_t)bh * Ls + q] = S;
    }
  }
}

// --------------- attention pass 2: weights out + O = w @ V -----------------
__global__ __launch_bounds__(256) void hypatt_apply(
    const float* __restrict__ Qp, const float* __restrict__ Kp,
    const float* __restrict__ Vp,
    const float* __restrict__ QSQ, const float* __restrict__ KSQ,
    const float* __restrict__ rowM, const float* __restrict__ rowS,
    float* __restrict__ wout,            // (B,H,L,L) weights output
    _Float16* __restrict__ attn_h)       // (B,L,D) f16 for o-projection
{
  __shared__ float kbuf[2][16 * KP];
  __shared__ float vbuf[2][16 * KP];
  __shared__ float wlds[8][16 * WP];     // padded 16x16 w-tile per wave

  const int wib  = threadIdx.x >> 5;
  const int wid  = blockIdx.x * 8 + wib;
  const int lane = threadIdx.x & 31;
  const int hf   = lane >> 4;
  const int lr   = lane & 15;
  const int bh   = wid >> 7;              // uniform across block
  const int qt   = wid & 127;

  const float* Qb  = Qp  + (size_t)bh * Ls * HD;
  const float* Kb  = Kp  + (size_t)bh * Ls * HD;
  const float* Vb  = Vp  + (size_t)bh * Ls * HD;
  const float* QSb = QSQ + (size_t)bh * Ls;
  const float* KSb = KSQ + (size_t)bh * Ls;
  float* wt = wlds[wib];

  v2f qa[16];
  const float* qrow = Qb + (size_t)(qt * 16 + lr) * HD + 2 * hf;
#pragma unroll
  for (int i = 0; i < 16; ++i) qa[i] = *(const v2f*)(qrow + 4 * i);

  float qs[8], icx[8], rm[8], ri[8];
#pragma unroll
  for (int j = 0; j < 8; ++j) {
    int q = qt * 16 + j + 8 * hf;
    qs[j]  = QSb[q];
    icx[j] = -4.0f / fmaxf(1.0f - qs[j], EPSF);
    rm[j]  = rowM[(size_t)bh * Ls + q];
    ri[j]  = 1.0f / rowS[(size_t)bh * Ls + q];
  }

  v8f oacc[4];
#pragma unroll
  for (int s = 0; s < 4; ++s) oacc[s] = zero8();

  stage_tile_16x64_f32(&kbuf[0][0], Kb);
  stage_tile_16x64_f32(&vbuf[0][0], Vb);

  for (int kt = 0; kt < KT; ++kt) {
    if (kt + 1 < KT) {
      stage_tile_16x64_f32(&kbuf[(kt + 1) & 1][0], Kb + (size_t)(kt + 1) * 16 * HD);
      stage_tile_16x64_f32(&vbuf[(kt + 1) & 1][0], Vb + (size_t)(kt + 1) * 16 * HD);
      stage_wait<2>();
    } else {
      stage_wait<0>();
    }
    __syncthreads();

    // -------- operand gather phase --------
    float ks  = KSb[kt * 16 + lr];
    const float* krow = &kbuf[kt & 1][lr * KP + 2 * hf];
    const float* vb   = &vbuf[kt & 1][0];
    v2f bb[16];
#pragma unroll
    for (int i = 0; i < 16; ++i) bb[i] = *(const v2f*)(krow + 4 * i);
    SCHED_FENCE();

    // -------- scores: two independent WMMA chains --------
    v8f xy0 = zero8(), xy1 = zero8();
#pragma unroll
    for (int i = 0; i < 8; ++i) {
      xy0 = __builtin_amdgcn_wmma_f32_16x16x4_f32(
          false, qa[2 * i],     false, bb[2 * i],     (short)0, xy0, false, false);
      xy1 = __builtin_amdgcn_wmma_f32_16x16x4_f32(
          false, qa[2 * i + 1], false, bb[2 * i + 1], (short)0, xy1, false, false);
    }

    // normalized weights into the per-wave LDS tile (C layout -> row major)
    float icy = 1.0f / fmaxf(1.0f - ks, EPSF);
#pragma unroll
    for (int j = 0; j < 8; ++j) {
      float xy  = xy0[j] + xy1[j];
      float dsq = fmaxf(qs[j] + ks - 2.0f * xy, 0.0f);
      float w   = __expf(dsq * icx[j] * icy - rm[j]) * ri[j];
      wt[(j + 8 * hf) * WP + lr] = w;
    }
    __builtin_amdgcn_wave_barrier();   // order LDS stores before reads

    // coalesced b128 write-back of the 16x16 weights tile (2 x float4/lane)
    {
      const int r = lane >> 1, c = (lane & 1) * 8;
      float4 w0 = *(const float4*)&wt[r * WP + c];
      float4 w1 = *(const float4*)&wt[r * WP + c + 4];
      float* wdst = wout + ((size_t)bh * Ls + qt * 16 + r) * Ls + kt * 16 + c;
      *(float4*)(wdst)     = w0;
      *(float4*)(wdst + 4) = w1;
    }

    // operand gather for O += w @ V : wa from wt, V scalars from vbuf
    v2f wa[4];
#pragma unroll
    for (int i = 0; i < 4; ++i)
      wa[i] = *(const v2f*)&wt[lr * WP + 4 * i + 2 * hf];
    float r0[4][4], r1[4][4];
#pragma unroll
    for (int i = 0; i < 4; ++i) {
      const int kk = 4 * i + 2 * hf;
#pragma unroll
      for (int sub = 0; sub < 4; ++sub) {
        r0[i][sub] = vb[kk * KP + sub * 16 + lr];
        r1[i][sub] = vb[(kk + 1) * KP + sub * 16 + lr];
      }
    }
    SCHED_FENCE();

    // O += w(16x16) @ V(16x64): 4 independent oacc chains, K=16 in 4 steps
#pragma unroll
    for (int i = 0; i < 4; ++i) {
#pragma unroll
      for (int sub = 0; sub < 4; ++sub) {
        v2f bbv; bbv.x = r0[i][sub]; bbv.y = r1[i][sub];
        oacc[sub] = __builtin_amdgcn_wmma_f32_16x16x4_f32(
            false, wa[i], false, bbv, (short)0, oacc[sub], false, false);
      }
    }
    __builtin_amdgcn_wave_barrier();   // keep next iter's wt stores after reads
    __syncthreads();
  }

  const int b = bh >> 4, h = bh & 15;
#pragma unroll
  for (int j = 0; j < 8; ++j) {
    int l = qt * 16 + j + 8 * hf;
    _Float16* o = attn_h + ((size_t)b * Ls + l) * Dd + h * 64 + lr;
#pragma unroll
    for (int sub = 0; sub < 4; ++sub) o[sub * 16] = (_Float16)oacc[sub][j];
  }
}

// ------------------------------- launcher ----------------------------------
extern "C" void kernel_launch(void* const* d_in, const int* in_sizes, int n_in,
                              void* d_out, int out_size, void* d_ws, size_t ws_size,
                              hipStream_t stream) {
  (void)in_sizes; (void)n_in; (void)out_size; (void)ws_size;

  const float* query = (const float*)d_in[0];
  const float* key   = (const float*)d_in[1];
  const float* value = (const float*)d_in[2];
  const float* q_w = (const float*)d_in[3];
  const float* q_b = (const float*)d_in[4];
  const float* k_w = (const float*)d_in[5];
  const float* k_b = (const float*)d_in[6];
  const float* v_w = (const float*)d_in[7];
  const float* v_b = (const float*)d_in[8];
  const float* o_w = (const float*)d_in[9];
  const float* o_b = (const float*)d_in[10];

  float* out0 = (float*)d_out;                       // (B,L,D)
  float* wout = out0 + (size_t)MTOT * Dd;            // (B,H,L,L)

  char* p = (char*)d_ws;
  auto carve = [&](size_t bytes) { void* r = (void*)p; p += (bytes + 255) & ~(size_t)255; return r; };
  float*    Qp   = (float*)   carve((size_t)Bb * Hh * Ls * HD * 4);
  float*    Kp   = (float*)   carve((size_t)Bb * Hh * Ls * HD * 4);
  float*    Vp   = (float*)   carve((size_t)Bb * Hh * Ls * HD * 4);
  _Float16* Xh   = (_Float16*)carve((size_t)MTOT * Dd * 2);
  _Float16* Wh   = (_Float16*)carve((size_t)Dd * Dd * 2);
  float*    QSQ  = (float*)   carve((size_t)Bb * Hh * Ls * 4);
  float*    KSQ  = (float*)   carve((size_t)Bb * Hh * Ls * 4);
  float*    rowM = (float*)   carve((size_t)Bb * Hh * Ls * 4);
  float*    rowS = (float*)   carve((size_t)Bb * Hh * Ls * 4);

  const int nX = MTOT * Dd;      // 8388608
  const int nW = Dd * Dd;        // 1048576
  const int gemmBlocks = (MTOT / 16) * (Dd / 64) / 8;   // 1024
  const int attnBlocks = Bb * Hh * (Ls / 16) / 8;       // 1024

  // Q projection (+ball, +|q|^2)
  hypatt_cvt_f16<<<nX / 256, 256, 0, stream>>>(query, Xh, nX);
  hypatt_cvt_f16<<<nW / 256, 256, 0, stream>>>(q_w, Wh, nW);
  hypatt_gemm<2><<<gemmBlocks, 256, 0, stream>>>(Xh, Wh, q_b, Qp, QSQ);

  // K projection (+ball, +|k|^2)
  hypatt_cvt_f16<<<nX / 256, 256, 0, stream>>>(key, Xh, nX);
  hypatt_cvt_f16<<<nW / 256, 256, 0, stream>>>(k_w, Wh, nW);
  hypatt_gemm<2><<<gemmBlocks, 256, 0, stream>>>(Xh, Wh, k_b, Kp, KSQ);

  // V projection
  hypatt_cvt_f16<<<nX / 256, 256, 0, stream>>>(value, Xh, nX);
  hypatt_cvt_f16<<<nW / 256, 256, 0, stream>>>(v_w, Wh, nW);
  hypatt_gemm<1><<<gemmBlocks, 256, 0, stream>>>(Xh, Wh, v_b, Vp, nullptr);

  // Attention
  hypatt_stats<<<attnBlocks, 256, 0, stream>>>(Qp, Kp, QSQ, KSQ, rowM, rowS);
  hypatt_apply<<<attnBlocks, 256, 0, stream>>>(Qp, Kp, Vp, QSQ, KSQ, rowM, rowS,
                                               wout, Xh /* reuse as attn_h */);

  // Output projection
  hypatt_cvt_f16<<<nW / 256, 256, 0, stream>>>(o_w, Wh, nW);
  hypatt_gemm<0><<<gemmBlocks, 256, 0, stream>>>(Xh, Wh, o_b, out0, nullptr);
}
